// ResGCN_29051158790634
// MI455X (gfx1250) — compile-verified, matrix operand order checked by reading, not run on hardware
//
#include <hip/hip_runtime.h>

#define NN 20000
#define EE 160000
#define PP 262144   // pow2 >= EE for bitonic sort
#define BB 16
#define NEGS 0.01f

typedef __bf16 bf16;
typedef __attribute__((ext_vector_type(16))) __bf16 v16bf;
typedef __attribute__((ext_vector_type(8)))  float  v8f;
typedef unsigned long long u64;

struct alignas(16) U4 { unsigned x, y, z, w; };
union FragA { v16bf v; U4 u[2]; };

__device__ inline v16bf load_frag(const bf16* p) {
  // elements 0..7  = K kb..kb+7   (VGPR0..3)
  // elements 8..15 = K kb+16..kb+23 (VGPR4..7)  per CDNA5 16-bit A/B striping
  FragA f;
  f.u[0] = *reinterpret_cast<const U4*>(p);
  f.u[1] = *reinterpret_cast<const U4*>(p + 16);
  return f.v;
}

__device__ inline float leakyf(float y) { return y >= 0.f ? y : NEGS * y; }

__device__ inline unsigned ordf(float f) {
  unsigned u = __float_as_uint(f);
  return (u & 0x80000000u) ? ~u : (u | 0x80000000u);
}
__device__ inline float unordf(unsigned s) {
  unsigned u = (s & 0x80000000u) ? (s & 0x7fffffffu) : ~s;
  return __uint_as_float(u);
}

// ---------------- utility kernels ----------------
__global__ void k_zero_f32(float* p, int n) {
  int i = blockIdx.x * blockDim.x + threadIdx.x;
  if (i < n) p[i] = 0.f;
}
__global__ void k_zero_u32(unsigned* p, int n) {
  int i = blockIdx.x * blockDim.x + threadIdx.x;
  if (i < n) p[i] = 0u;
}
__global__ void k_f32_bf16(const float* __restrict__ in, bf16* __restrict__ out, int n) {
  int i = blockIdx.x * blockDim.x + threadIdx.x;
  if (i < n) out[i] = (bf16)in[i];
}
__global__ void k_init_edges(const int* __restrict__ ei, int* es, int* ed, int* ev) {
  int i = blockIdx.x * blockDim.x + threadIdx.x;
  if (i < EE) { es[i] = ei[i]; ed[i] = ei[EE + i]; ev[i] = 1; }
}
__global__ void k_init_ones_i32(int* p, int n) {
  int i = blockIdx.x * blockDim.x + threadIdx.x;
  if (i < n) p[i] = 1;
}

// ---------------- SAGE aggregation ----------------
__global__ void k_scatter_mean(const float* __restrict__ x, const int* __restrict__ src,
                               const int* __restrict__ dst, const int* __restrict__ ev,
                               float* __restrict__ agg, float* __restrict__ cnt, int C) {
  int i = blockIdx.x * blockDim.x + threadIdx.x;
  if (i >= EE * C) return;
  int e = i / C, c = i - e * C;
  if (!ev[e]) return;
  int s = src[e], d = dst[e];
  atomicAdd(&agg[(size_t)d * C + c], x[(size_t)s * C + c]);
  if (c == 0) atomicAdd(&cnt[d], 1.f);
}
__global__ void k_mean_div(float* __restrict__ agg, const float* __restrict__ cnt, int C) {
  int i = blockIdx.x * blockDim.x + threadIdx.x;
  if (i >= NN * C) return;
  int r = i / C;
  agg[i] /= fmaxf(cnt[r], 1.f);
}

// ---------------- WMMA dual GEMM:  out = A1@W1^T + A2@W2^T + bias -------------
// A: [M x K] bf16 row-major, W: [Co x K] bf16 row-major (i.e. B = W^T), out f32.
// One wave per 16x16 output tile; weights & activations are L2-resident (192MB L2)
// so fragments are loaded straight from global, no LDS staging needed.
__global__ void k_wmma_gemm_dual(const bf16* __restrict__ A1, const bf16* __restrict__ W1,
                                 const bf16* __restrict__ A2, const bf16* __restrict__ W2,
                                 const float* __restrict__ bias, float* __restrict__ out,
                                 int M, int K, int Co) {
  int wave = threadIdx.x >> 5;
  int lane = threadIdx.x & 31;
  int tiles_n = Co >> 4;
  int tile = blockIdx.x * (blockDim.x >> 5) + wave;
  if (tile >= (M >> 4) * tiles_n) return;      // wave-uniform: EXEC stays all-ones
  int tm = tile / tiles_n, tn = tile - tm * tiles_n;
  int r16 = lane & 15;
  int kb  = (lane >> 4) << 3;                  // lanes 0-15 -> K base 0, 16-31 -> 8

  v8f c;
#pragma unroll
  for (int i = 0; i < 8; ++i) c[i] = 0.f;

  const bf16* arow = A1 + (size_t)(tm * 16 + r16) * K + kb;
  const bf16* brow = W1 + (size_t)(tn * 16 + r16) * K + kb;
  for (int kk = 0; kk < K; kk += 32) {
    v16bf a = load_frag(arow + kk);
    v16bf b = load_frag(brow + kk);
    c = __builtin_amdgcn_wmma_f32_16x16x32_bf16(false, a, false, b, (short)0, c, false, false);
  }
  if (A2) {
    const bf16* arow2 = A2 + (size_t)(tm * 16 + r16) * K + kb;
    const bf16* brow2 = W2 + (size_t)(tn * 16 + r16) * K + kb;
    for (int kk = 0; kk < K; kk += 32) {
      v16bf a = load_frag(arow2 + kk);
      v16bf b = load_frag(brow2 + kk);
      c = __builtin_amdgcn_wmma_f32_16x16x32_bf16(false, a, false, b, (short)0, c, false, false);
    }
  }
  float bv = bias ? bias[tn * 16 + r16] : 0.f;
  int rbase = tm * 16 + ((lane >> 4) << 3);    // C/D layout: lanes 0-15 rows r, 16-31 rows r+8
  int col   = tn * 16 + r16;
#pragma unroll
  for (int r = 0; r < 8; ++r)
    out[(size_t)(rbase + r) * Co + col] = c[r] + bv;
}

// ---------------- L2 row-normalize (wave per row, wave32 shuffles) ------------
__global__ void k_l2norm(float* __restrict__ x, int C) {
  int wave = threadIdx.x >> 5, lane = threadIdx.x & 31;
  int row = blockIdx.x * (blockDim.x >> 5) + wave;
  if (row >= NN) return;
  float s = 0.f;
  for (int c = lane; c < C; c += 32) { float v = x[(size_t)row * C + c]; s += v * v; }
  for (int m = 16; m; m >>= 1) s += __shfl_xor(s, m, 32);
  float nrm = fmaxf(sqrtf(s), 1e-12f);
  for (int c = lane; c < C; c += 32) x[(size_t)row * C + c] /= nrm;
}

// ---------------- BatchNorm (masked batch statistics) ----------------
__global__ void k_count_valid(const int* __restrict__ nv, float* vcnt, int n) {
  int i = blockIdx.x * blockDim.x + threadIdx.x;
  if (i < n && nv[i]) atomicAdd(vcnt, 1.f);
}
__global__ void k_bn_stats(const float* __restrict__ x, const int* __restrict__ nv,
                           float* __restrict__ s, float* __restrict__ q, int C, int rpb) {
  int c = blockIdx.y * blockDim.x + threadIdx.x;
  if (c >= C) return;
  int r0 = blockIdx.x * rpb;
  int r1 = r0 + rpb; if (r1 > NN) r1 = NN;
  float ls = 0.f, lq = 0.f;
  for (int r = r0; r < r1; ++r)
    if (nv[r]) { float v = x[(size_t)r * C + c]; ls += v; lq += v * v; }
  atomicAdd(&s[c], ls);
  atomicAdd(&q[c], lq);
}
__global__ void k_bn_apply(const float* __restrict__ x, float* __restrict__ y,
                           const float* __restrict__ s, const float* __restrict__ q,
                           const float* __restrict__ vcnt, const float* __restrict__ g,
                           const float* __restrict__ b, const int* __restrict__ nv,
                           int C, int mode) {
  int i = blockIdx.x * blockDim.x + threadIdx.x;
  if (i >= NN * C) return;
  int r = i / C, c = i - r * C;
  float cnt = fmaxf(vcnt[0], 1.f);
  float m = s[c] / cnt;
  float v = q[c] / cnt - m * m;
  float t = (x[i] - m) * rsqrtf(v + 1e-5f) * g[c] + b[c];
  if (mode == 1) { t = leakyf(t); t *= nv[r] ? 1.f : 0.f; }
  y[i] = t;
}
__global__ void k_residual(float* __restrict__ h, const float* __restrict__ t,
                           const int* __restrict__ nv, int C) {
  int i = blockIdx.x * blockDim.x + threadIdx.x;
  if (i >= NN * C) return;
  int r = i / C;
  h[i] = (h[i] + t[i]) * (nv[r] ? 1.f : 0.f);
}

// ---------------- EdgePooling ----------------
// score kernel: weight vector staged into LDS via async global->LDS path
__global__ void k_edge_score(const float* __restrict__ x, const int* __restrict__ src,
                             const int* __restrict__ dst, const int* __restrict__ ev,
                             const float* __restrict__ w, const float* __restrict__ bsc,
                             float* __restrict__ raw, int C) {
  __shared__ float sw[1024];
  int twoC = 2 * C;
  for (int t = threadIdx.x; t < twoC; t += blockDim.x) {
    unsigned loff = (unsigned)(unsigned long long)(&sw[t]);         // LDS byte offset (flat addr low bits)
    unsigned long long ga = (unsigned long long)(const void*)(w + t);
    asm volatile("global_load_async_to_lds_b32 %0, %1, off" :: "v"(loff), "v"(ga) : "memory");
  }
  asm volatile("s_wait_asynccnt 0" ::: "memory");
  __syncthreads();
  int wave = threadIdx.x >> 5, lane = threadIdx.x & 31;
  int e = blockIdx.x * (blockDim.x >> 5) + wave;
  if (e >= EE) return;
  int s = src[e], d = dst[e];
  float acc = 0.f;
  for (int c = lane; c < C; c += 32)
    acc += sw[c] * x[(size_t)s * C + c] + sw[C + c] * x[(size_t)d * C + c];
  for (int m = 16; m; m >>= 1) acc += __shfl_xor(acc, m, 32);
  if (lane == 0) raw[e] = ev[e] ? (acc + bsc[0]) : -1e9f;
}
__global__ void k_seg_max(const float* __restrict__ raw, const int* __restrict__ dst,
                          const int* __restrict__ ev, unsigned* __restrict__ mxu) {
  int i = blockIdx.x * blockDim.x + threadIdx.x;
  if (i >= EE || !ev[i]) return;
  atomicMax(&mxu[dst[i]], ordf(raw[i]));
}
__global__ void k_seg_expden(const float* __restrict__ raw, const int* __restrict__ dst,
                             const int* __restrict__ ev, const unsigned* __restrict__ mxu,
                             float* __restrict__ den, float* __restrict__ exb) {
  int i = blockIdx.x * blockDim.x + threadIdx.x;
  if (i >= EE) return;
  if (!ev[i]) { exb[i] = 0.f; return; }
  int d = dst[i];
  float e_ = __expf(raw[i] - unordf(mxu[d]));
  exb[i] = e_;
  atomicAdd(&den[d], e_);
}
__global__ void k_score_final(const float* __restrict__ exb, const int* __restrict__ dst,
                              const int* __restrict__ ev, const float* __restrict__ den,
                              float* __restrict__ score) {
  int i = blockIdx.x * blockDim.x + threadIdx.x;
  if (i >= EE) return;
  score[i] = ev[i] ? (exb[i] / fmaxf(den[dst[i]], 1e-16f) + 0.5f) : 0.f;
}
// stable descending-score sort key: hi = ~ord(score), lo = index
__global__ void k_sortkeys_score(const float* __restrict__ score, const int* __restrict__ ev,
                                 u64* __restrict__ keys) {
  int i = blockIdx.x * blockDim.x + threadIdx.x;
  if (i >= PP) return;
  if (i < EE) {
    float s = ev[i] ? score[i] : -__builtin_inff();
    keys[i] = ((u64)(~ordf(s)) << 32) | (unsigned)i;
  } else {
    keys[i] = ~0ull;
  }
}
__global__ void k_bitonic(u64* __restrict__ keys, unsigned j, unsigned k) {
  unsigned i = blockIdx.x * blockDim.x + threadIdx.x;
  unsigned ixj = i ^ j;
  if (ixj > i) {
    u64 a = keys[i], b = keys[ixj];
    bool up = ((i & k) == 0);
    if ((a > b) == up) { keys[i] = b; keys[ixj] = a; }
  }
}
__global__ void k_gather_sorted(const u64* __restrict__ keys, const int* __restrict__ es,
                                const int* __restrict__ ed, const int* __restrict__ ev,
                                const float* __restrict__ sc, int* __restrict__ ss,
                                int* __restrict__ sd, int* __restrict__ se,
                                float* __restrict__ ssc) {
  int i = blockIdx.x * blockDim.x + threadIdx.x;
  if (i >= EE) return;
  unsigned idx = (unsigned)(keys[i] & 0xffffffffu);
  ss[i] = es[idx]; sd[i] = ed[idx]; se[i] = ev[idx]; ssc[i] = sc[idx];
}
// sequential greedy matching; marked[] lives in LDS (20 KB of the 320 KB WGP LDS)
__global__ void k_greedy(const int* __restrict__ ss, const int* __restrict__ sd,
                         const int* __restrict__ se, const int* __restrict__ nv,
                         int* __restrict__ chosen) {
  __shared__ char marked[NN];
  for (int i = threadIdx.x; i < NN; i += blockDim.x) marked[i] = nv[i] ? 0 : 1;
  __syncthreads();
  if (threadIdx.x == 0) {
    for (int e = 0; e < EE; ++e) {
      int s = ss[e], d = sd[e];
      int can = se[e] && !marked[s] && !marked[d];
      if (can) { marked[s] = 1; marked[d] = 1; }
      chosen[e] = can;
    }
  }
}
__global__ void k_init_cluster(int* __restrict__ cluster, float* __restrict__ sn) {
  int i = blockIdx.x * blockDim.x + threadIdx.x;
  if (i < NN) { cluster[i] = i; sn[i] = 1.f; }
}
__global__ void k_apply_chosen(const int* __restrict__ ss, const int* __restrict__ sd,
                               const float* __restrict__ ssc, const int* __restrict__ chosen,
                               int* __restrict__ cluster, float* __restrict__ sn) {
  int e = blockIdx.x * blockDim.x + threadIdx.x;
  if (e >= EE || !chosen[e]) return;
  int s = ss[e], d = sd[e];
  int rep = s < d ? s : d;
  int oth = s < d ? d : s;
  cluster[oth] = rep;
  sn[rep] = ssc[e];
}
__global__ void k_newnv(const int* __restrict__ nv, const int* __restrict__ cluster,
                        int* __restrict__ nv2) {
  int i = blockIdx.x * blockDim.x + threadIdx.x;
  if (i < NN) nv2[i] = (nv[i] && cluster[i] == i) ? 1 : 0;
}
__global__ void k_pool_scatter(const float* __restrict__ h, const int* __restrict__ cluster,
                               float* __restrict__ nx, int C) {
  int i = blockIdx.x * blockDim.x + threadIdx.x;
  if (i >= NN * C) return;
  int r = i / C, c = i - r * C;
  atomicAdd(&nx[(size_t)cluster[r] * C + c], h[i]);
}
__global__ void k_pool_scale(float* __restrict__ nx, const float* __restrict__ sn,
                             const int* __restrict__ nv2, int C) {
  int i = blockIdx.x * blockDim.x + threadIdx.x;
  if (i >= NN * C) return;
  int r = i / C;
  nx[i] *= sn[r] * (nv2[r] ? 1.f : 0.f);
}
__global__ void k_remap_keys(const int* __restrict__ es, const int* __restrict__ ed,
                             const int* __restrict__ ev, const int* __restrict__ cluster,
                             int* __restrict__ tns, int* __restrict__ tnd,
                             u64* __restrict__ keys) {
  int i = blockIdx.x * blockDim.x + threadIdx.x;
  if (i >= PP) return;
  if (i < EE) {
    int ns = cluster[es[i]], nd = cluster[ed[i]];
    tns[i] = ns; tnd[i] = nd;
    unsigned hi = ev[i] ? (unsigned)(ns * (NN + 1) + nd) : (unsigned)((NN + 1) * (u64)(NN + 1));
    keys[i] = ((u64)hi << 32) | (unsigned)i;
  } else {
    keys[i] = ~0ull;
  }
}
__global__ void k_coalesce(const u64* __restrict__ keys, const int* __restrict__ tns,
                           const int* __restrict__ tnd, const int* __restrict__ evold,
                           int* __restrict__ es2, int* __restrict__ ed2, int* __restrict__ ev2) {
  int i = blockIdx.x * blockDim.x + threadIdx.x;
  if (i >= EE) return;
  u64 kc = keys[i];
  unsigned idx = (unsigned)(kc & 0xffffffffu);
  unsigned hi = (unsigned)(kc >> 32);
  int first = (i == 0) || (hi != (unsigned)(keys[i - 1] >> 32));
  es2[i] = tns[idx];
  ed2[i] = tnd[idx];
  ev2[i] = (first && evold[idx]) ? 1 : 0;
}

// ---------------- readout ----------------
__global__ void k_batch_scatter(const float* __restrict__ h, const int* __restrict__ batch,
                                float* __restrict__ g, int C) {
  int i = blockIdx.x * blockDim.x + threadIdx.x;
  if (i >= NN * C) return;
  int r = i / C, c = i - r * C;
  atomicAdd(&g[(size_t)batch[r] * C + c], h[i]);
}
__global__ void k_bn_final(const float* __restrict__ x, const float* __restrict__ g,
                           const float* __restrict__ b, float* __restrict__ out) {
  int c = threadIdx.x;  // 256 channels
  float m = 0.f;
  for (int r = 0; r < BB; ++r) m += x[r * 256 + c];
  m /= (float)BB;
  float v = 0.f;
  for (int r = 0; r < BB; ++r) { float d = x[r * 256 + c] - m; v += d * d; }
  v /= (float)BB;
  float sc = rsqrtf(v + 1e-5f) * g[c];
  for (int r = 0; r < BB; ++r) {
    float y = (x[r * 256 + c] - m) * sc + b[c];
    out[r * 256 + c] = leakyf(y);
  }
}

// =====================================================================
extern "C" void kernel_launch(void* const* d_in, const int* in_sizes, int n_in,
                              void* d_out, int out_size, void* d_ws, size_t ws_size,
                              hipStream_t stream) {
  (void)in_sizes; (void)n_in; (void)out_size; (void)ws_size;
  auto F = [&](int i) { return (const float*)d_in[i]; };
  auto I = [&](int i) { return (const int*)d_in[i]; };

  // ---- workspace carve-up ----
  char* base = (char*)d_ws;
  size_t off = 0;
  auto alloc = [&](size_t bytes) -> void* {
    size_t a = (off + 255) & ~(size_t)255;
    off = a + bytes;
    return (void*)(base + a);
  };
  float* fH   = (float*)alloc((size_t)NN * 512 * 4);
  float* fT   = (float*)alloc((size_t)NN * 512 * 4);
  float* fO   = (float*)alloc((size_t)NN * 512 * 4);
  float* fM   = (float*)alloc((size_t)NN * 512 * 4);
  float* fCnt = (float*)alloc((size_t)NN * 4);
  bf16*  bA   = (bf16*)alloc((size_t)NN * 512 * 2);
  bf16*  bX   = (bf16*)alloc((size_t)NN * 512 * 2);
  bf16*  bW1  = (bf16*)alloc((size_t)512 * 512 * 2);
  bf16*  bW2  = (bf16*)alloc((size_t)512 * 512 * 2);
  float* chS  = (float*)alloc(512 * 4);
  float* chQ  = (float*)alloc(512 * 4);
  float* vcnt = (float*)alloc(256);
  float* raw  = (float*)alloc((size_t)EE * 4);
  float* exb  = (float*)alloc((size_t)EE * 4);
  float* scor = (float*)alloc((size_t)EE * 4);
  unsigned* mxu = (unsigned*)alloc((size_t)(NN + 1) * 4);
  float* den  = (float*)alloc((size_t)(NN + 1) * 4);
  u64*   keys = (u64*)alloc((size_t)PP * 8);
  int* ssrc = (int*)alloc((size_t)EE * 4);
  int* sdst = (int*)alloc((size_t)EE * 4);
  int* sev  = (int*)alloc((size_t)EE * 4);
  float* ssc = (float*)alloc((size_t)EE * 4);
  int* chosen = (int*)alloc((size_t)EE * 4);
  int* cluster = (int*)alloc((size_t)NN * 4);
  float* snv = (float*)alloc((size_t)NN * 4);
  int* nvA = (int*)alloc((size_t)NN * 4);
  int* nvB = (int*)alloc((size_t)NN * 4);
  int* esA = (int*)alloc((size_t)EE * 4);
  int* edA = (int*)alloc((size_t)EE * 4);
  int* evA = (int*)alloc((size_t)EE * 4);
  int* esB = (int*)alloc((size_t)EE * 4);
  int* edB = (int*)alloc((size_t)EE * 4);
  int* evB = (int*)alloc((size_t)EE * 4);
  int* tns = (int*)alloc((size_t)EE * 4);
  int* tnd = (int*)alloc((size_t)EE * 4);
  float* gbuf = (float*)alloc((size_t)BB * 512 * 4);

  float *pH = fH, *pM = fM;
  int *pes = esA, *ped = edA, *pev = evA, *pes2 = esB, *ped2 = edB, *pev2 = evB;
  int *pnv = nvA, *pnv2 = nvB;

  auto CB = [](int n) { return (unsigned)((n + 255) / 256); };
  auto Z  = [&](float* p, int n) { k_zero_f32<<<CB(n), 256, 0, stream>>>(p, n); };
  auto Zu = [&](unsigned* p, int n) { k_zero_u32<<<CB(n), 256, 0, stream>>>(p, n); };
  auto CV = [&](const float* in, bf16* out, int n) { k_f32_bf16<<<CB(n), 256, 0, stream>>>(in, out, n); };

  auto gemm = [&](const bf16* A1, const bf16* W1, const bf16* A2, const bf16* W2,
                  const float* bias, float* out, int M, int K, int Co) {
    int tiles = (M >> 4) * (Co >> 4);
    k_wmma_gemm_dual<<<(tiles + 3) / 4, 128, 0, stream>>>(A1, W1, A2, W2, bias, out, M, K, Co);
  };
  auto bitonic = [&](u64* k) {
    for (unsigned kk = 2; kk <= PP; kk <<= 1)
      for (unsigned j = kk >> 1; j > 0; j >>= 1)
        k_bitonic<<<PP / 256, 256, 0, stream>>>(k, j, kk);
  };
  auto sage = [&](const float* xin, int Cin, int Cout, const float* Wl, const float* bl,
                  const float* Wr, float* outF) {
    Z(pM, NN * Cin);
    Z(fCnt, NN);
    k_scatter_mean<<<CB(EE * Cin), 256, 0, stream>>>(xin, pes, ped, pev, pM, fCnt, Cin);
    k_mean_div<<<CB(NN * Cin), 256, 0, stream>>>(pM, fCnt, Cin);
    CV(pM, bA, NN * Cin);
    CV(xin, bX, NN * Cin);
    CV(Wl, bW1, Cout * Cin);
    CV(Wr, bW2, Cout * Cin);
    gemm(bA, bW1, bX, bW2, bl, outF, NN, Cin, Cout);
    k_l2norm<<<(NN + 7) / 8, 256, 0, stream>>>(outF, Cout);
  };
  auto bnrun = [&](const float* xin, float* yout, const float* g, const float* b, int C, int mode) {
    Z(chS, C); Z(chQ, C); Z(vcnt, 1);
    k_count_valid<<<CB(NN), 256, 0, stream>>>(pnv, vcnt, NN);
    dim3 gs((NN + 127) / 128, (C + 255) / 256);
    k_bn_stats<<<gs, 256, 0, stream>>>(xin, pnv, chS, chQ, C, 128);
    k_bn_apply<<<CB(NN * C), 256, 0, stream>>>(xin, yout, chS, chQ, vcnt, g, b, pnv, C, mode);
  };
  auto stage = [&](const float* xin, int Cin, int Cout, const float* sWl, const float* sbl,
                   const float* sWr, const float* bg, const float* bb, const float* lWl,
                   const float* lbl, const float* lWr, const float* lg, const float* lb) {
    sage(xin, Cin, Cout, sWl, sbl, sWr, fO);
    bnrun(fO, pH, bg, bb, Cout, 1);
    for (int t = 0; t < 2; ++t) {   // DeepGCN res+ x TIME, act = identity (slope 1.0)
      bnrun(pH, fT, lg, lb, Cout, 0);
      sage(fT, Cout, Cout, lWl, lbl, lWr, fO);
      k_residual<<<CB(NN * Cout), 256, 0, stream>>>(pH, fO, pnv, Cout);
    }
  };
  auto pool = [&](int C, const float* w, const float* bsc) {
    k_edge_score<<<EE / 8, 256, 0, stream>>>(pH, pes, ped, pev, w, bsc, raw, C);
    Zu(mxu, NN + 1);
    k_seg_max<<<CB(EE), 256, 0, stream>>>(raw, ped, pev, mxu);
    Z(den, NN + 1);
    k_seg_expden<<<CB(EE), 256, 0, stream>>>(raw, ped, pev, mxu, den, exb);
    k_score_final<<<CB(EE), 256, 0, stream>>>(exb, ped, pev, den, scor);
    k_sortkeys_score<<<PP / 256, 256, 0, stream>>>(scor, pev, keys);
    bitonic(keys);
    k_gather_sorted<<<CB(EE), 256, 0, stream>>>(keys, pes, ped, pev, scor, ssrc, sdst, sev, ssc);
    k_greedy<<<1, 1024, 0, stream>>>(ssrc, sdst, sev, pnv, chosen);
    k_init_cluster<<<CB(NN), 256, 0, stream>>>(cluster, snv);
    k_apply_chosen<<<CB(EE), 256, 0, stream>>>(ssrc, sdst, ssc, chosen, cluster, snv);
    k_newnv<<<CB(NN), 256, 0, stream>>>(pnv, cluster, pnv2);
    Z(pM, NN * C);
    k_pool_scatter<<<CB(NN * C), 256, 0, stream>>>(pH, cluster, pM, C);
    k_pool_scale<<<CB(NN * C), 256, 0, stream>>>(pM, snv, pnv2, C);
    k_remap_keys<<<PP / 256, 256, 0, stream>>>(pes, ped, pev, cluster, tns, tnd, keys);
    bitonic(keys);
    k_coalesce<<<CB(EE), 256, 0, stream>>>(keys, tns, tnd, pev, pes2, ped2, pev2);
    { float* t_ = pH; pH = pM; pM = t_; }
    { int* t_ = pes; pes = pes2; pes2 = t_; }
    { int* t_ = ped; ped = ped2; ped2 = t_; }
    { int* t_ = pev; pev = pev2; pev2 = t_; }
    { int* t_ = pnv; pnv = pnv2; pnv2 = t_; }
  };

  // ---- init ----
  k_init_edges<<<CB(EE), 256, 0, stream>>>(I(1), pes, ped, pev);
  k_init_ones_i32<<<CB(NN), 256, 0, stream>>>(pnv, NN);

  // ---- 4 stages + 3 pools ----
  stage(F(0), 128, 256, F(3), F(4), F(5), F(6), F(7), F(8), F(9), F(10), F(11), F(12));
  pool(256, F(13), F(14));
  stage(pH, 256, 256, F(15), F(16), F(17), F(18), F(19), F(20), F(21), F(22), F(23), F(24));
  pool(256, F(25), F(26));
  stage(pH, 256, 512, F(27), F(28), F(29), F(30), F(31), F(32), F(33), F(34), F(35), F(36));
  pool(512, F(37), F(38));
  stage(pH, 512, 512, F(39), F(40), F(41), F(42), F(43), F(44), F(45), F(46), F(47), F(48));

  // ---- readout: global_add_pool -> tr1 linear (WMMA) -> bn5 + leaky ----
  Z(gbuf, BB * 512);
  k_batch_scatter<<<CB(NN * 512), 256, 0, stream>>>(pH, I(2), gbuf, 512);
  CV(gbuf, bA, BB * 512);
  CV(F(51), bW1, 256 * 512);
  gemm(bA, bW1, nullptr, nullptr, F(52), fO, BB, 512, 256);
  k_bn_final<<<1, 256, 0, stream>>>(fO, F(49), F(50), (float*)d_out);
}
